// MoETopPLayer_20289425507149
// MI455X (gfx1250) — matrix operand order repaired and checked by reading, not compiled, not run on hardware
//
#include <hip/hip_runtime.h>
#include <hip/hip_bf16.h>
#include <math.h>

#define NEXP 8
#define DM   512
#define DFF  2048
#define NTOK 8192          // B*T = 4*2048
#define TOPP 0.9f
#define MT   64            // token tile per FFN block

typedef __attribute__((ext_vector_type(16))) _Float16 v16h;
typedef __attribute__((ext_vector_type(8)))  _Float16 v8h;
typedef __attribute__((ext_vector_type(8)))  float    v8f;
typedef __attribute__((ext_vector_type(4)))  int      v4i;

union V16 { v16h v; v8h h[2]; };

#if defined(__has_builtin)
#if __has_builtin(__builtin_amdgcn_global_load_async_to_lds_b128)
#define USE_ASYNC_LDS 1
#endif
#endif

// ---------------- conversion kernels (fp32 -> f16, weights transposed) ---------------

__global__ void cvt_x_kernel(const float* __restrict__ x, _Float16* __restrict__ xh, int n) {
    int i = blockIdx.x * blockDim.x + threadIdx.x;
    if (i < n) xh[i] = (_Float16)x[i];
}

// W1: [e][d][ff] fp32 -> W1t: [e][ff][d] f16  (K=d contiguous for WMMA B-fragments)
__global__ void cvt_w1_kernel(const float* __restrict__ W1, _Float16* __restrict__ W1t) {
    int i = blockIdx.x * blockDim.x + threadIdx.x;
    const int per = DM * DFF;
    if (i >= NEXP * per) return;
    int e = i / per, rem = i % per;
    int d = rem / DFF, f = rem % DFF;
    W1t[((size_t)e * DFF + f) * DM + d] = (_Float16)W1[i];
}

// W2: [e][ff][dout] fp32 -> W2t: [e][dout][ff] f16 (K=ff contiguous)
__global__ void cvt_w2_kernel(const float* __restrict__ W2, _Float16* __restrict__ W2t) {
    int i = blockIdx.x * blockDim.x + threadIdx.x;
    const int per = DFF * DM;
    if (i >= NEXP * per) return;
    int e = i / per, rem = i % per;
    int f = rem / DM, dd = rem % DM;
    W2t[((size_t)e * DM + dd) * DFF + f] = (_Float16)W2[i];
}

// ---------------- router: softmax, top-4, top-p keep, gather lists, aux stats --------

__global__ __launch_bounds__(256)
void router_kernel(const float* __restrict__ x, const float* __restrict__ rw,
                   int* __restrict__ tok_list, float* __restrict__ w_list,
                   int* __restrict__ cnt, float* __restrict__ Psum) {
    const int wv = threadIdx.x >> 5, lane = threadIdx.x & 31;
    const int t = blockIdx.x * 8 + wv;            // one wave per token, 8 waves/block
    __shared__ float sP[NEXP];
    if (threadIdx.x < NEXP) sP[threadIdx.x] = 0.f;
    __syncthreads();

    float acc[NEXP];
#pragma unroll
    for (int e = 0; e < NEXP; ++e) acc[e] = 0.f;
    if (t < NTOK) {
#pragma unroll 4
        for (int i = 0; i < DM / 32; ++i) {
            int d = lane + i * 32;
            float xv = x[(size_t)t * DM + d];
#pragma unroll
            for (int e = 0; e < NEXP; ++e) acc[e] += xv * rw[d * NEXP + e];
        }
    }
#pragma unroll
    for (int e = 0; e < NEXP; ++e) {
        float v = acc[e];
        for (int off = 16; off > 0; off >>= 1) v += __shfl_down(v, off, 32);
        acc[e] = v;
    }

    if (lane == 0 && t < NTOK) {
        float mx = acc[0];
#pragma unroll
        for (int e = 1; e < NEXP; ++e) mx = fmaxf(mx, acc[e]);
        float p[NEXP], s = 0.f;
#pragma unroll
        for (int e = 0; e < NEXP; ++e) { p[e] = __expf(acc[e] - mx); s += p[e]; }
        float inv = 1.f / s;
#pragma unroll
        for (int e = 0; e < NEXP; ++e) { p[e] *= inv; atomicAdd(&sP[e], p[e]); }

        // top-4 selection (stable for ties: earliest index wins, matches lax.top_k)
        float sp[NEXP]; int si[NEXP];
#pragma unroll
        for (int e = 0; e < NEXP; ++e) { sp[e] = p[e]; si[e] = e; }
#pragma unroll
        for (int k = 0; k < 4; ++k) {
            int best = k;
#pragma unroll
            for (int j2 = k + 1; j2 < NEXP; ++j2) if (sp[j2] > sp[best]) best = j2;
            float tv = sp[k]; sp[k] = sp[best]; sp[best] = tv;
            int ti = si[k]; si[k] = si[best]; si[best] = ti;
        }
        // top-p keep: keep[k] = (k < MIN_EXPERTS) | (cumsum_before_k < 0.9)
        float cum = 0.f, w[4], wsum = 0.f; int kid[4];
#pragma unroll
        for (int k = 0; k < 4; ++k) {
            bool keep = (k == 0) || (cum < TOPP);
            cum += sp[k];
            w[k] = keep ? sp[k] : 0.f;
            kid[k] = keep ? si[k] : -1;
            wsum += w[k];
        }
        wsum = fmaxf(wsum, 1e-9f);
#pragma unroll
        for (int k = 0; k < 4; ++k) {
            if (kid[k] >= 0) {
                int pos = atomicAdd(&cnt[kid[k]], 1);
                tok_list[kid[k] * NTOK + pos] = t;
                w_list[kid[k] * NTOK + pos] = w[k] / wsum;
            }
        }
    }
    __syncthreads();
    if (threadIdx.x < NEXP) atomicAdd(&Psum[threadIdx.x], sP[threadIdx.x]);
}

// ---------------- aux loss finalize --------------------------------------------------

__global__ void aux_kernel(const int* __restrict__ cnt, const float* __restrict__ Psum,
                           float* __restrict__ out_aux) {
    float total = 0.f, c[NEXP];
#pragma unroll
    for (int e = 0; e < NEXP; ++e) { c[e] = (float)cnt[e]; total += c[e]; }
    total = fmaxf(total, 1.f);
    float aux = 0.f;
#pragma unroll
    for (int e = 0; e < NEXP; ++e) aux += (c[e] / total) * (Psum[e] / (float)NTOK);
    out_aux[0] = 0.01f * (float)NEXP * aux;
}

// ---------------- fused expert FFN: gelu(X@W1+b1)@W2 + b2, WMMA f16 ------------------
// Block = 512 threads = 16 waves, handles 64 gathered tokens of one expert.
// Wave wv: cw = wv&7 (column group), rg = wv>>3 (32-row pair).
// LDS: A tile 64x512 f16 (64KB, async-loaded) + H chunk 64x128 f16 (16KB).
// Phase A: wave computes two 16x16 H subtiles (rows rg*32..+31, cols cw*16) over K=512;
//          B-fragment loaded once, reused for both row tiles.
// Phase B: wave owns out cols [cw*64, cw*64+64) x rows rg*32..+31: 8 persistent
//          f32 accumulator tiles, K=128 per chunk, B reused across the row pair.

__global__ __launch_bounds__(512)
void ffn_kernel(const _Float16* __restrict__ xh, const _Float16* __restrict__ W1t,
                const _Float16* __restrict__ W2t, const float* __restrict__ b1,
                const float* __restrict__ b2, const int* __restrict__ tok_list,
                const float* __restrict__ w_list, const int* __restrict__ cnt,
                float* __restrict__ out) {
    const int e = blockIdx.y;
    const int ne = cnt[e];
    const int start = blockIdx.x * MT;
    if (start >= ne) return;                       // block-uniform exit, EXEC stays full

    __shared__ __align__(16) _Float16 Alds[MT * DM];    // 64 KB
    __shared__ __align__(16) _Float16 Hlds[MT * 128];   // 16 KB
    __shared__ int   stok[MT];
    __shared__ float swt[MT];

    const int tid = threadIdx.x;
    const int wv = tid >> 5, lane = tid & 31;
    const int hl = lane >> 4, ln = lane & 15;      // A/B half-wave split (K groups)
    const int cw = wv & 7;                         // column group
    const int rg = wv >> 3;                        // row-pair group (0/1)

    if (tid < MT) {
        int r = start + tid;
        stok[tid] = (r < ne) ? tok_list[e * NTOK + r] : 0;
        swt[tid]  = (r < ne) ? w_list[e * NTOK + r] : 0.f;  // pad rows get weight 0
    }
    __syncthreads();

    // stage gathered A rows into LDS: 64 rows x 1KB = 64KB
#ifdef USE_ASYNC_LDS
    {
#pragma unroll
        for (int it = 0; it < (MT * DM * 2) / (512 * 16); ++it) {   // 8 iters
            int bo = (it * 512 + tid) * 16;                          // byte offset
            int row = bo >> 10, within = bo & 1023;
            const char* src = (const char*)xh + ((size_t)stok[row] << 10) + within;
            char* dst = (char*)Alds + bo;
            __builtin_amdgcn_global_load_async_to_lds_b128((v4i*)src, (v4i*)dst, 0, 0);
        }
        asm volatile("s_wait_asynccnt 0" ::: "memory");
    }
#else
    {
#pragma unroll
        for (int it = 0; it < (MT * DM * 2) / (512 * 16); ++it) {
            int bo = (it * 512 + tid) * 16;
            int row = bo >> 10, within = bo & 1023;
            *(v8h*)((char*)Alds + bo) =
                *(const v8h*)((const char*)xh + ((size_t)stok[row] << 10) + within);
        }
    }
#endif
    __syncthreads();

    v8f oacc[4][2];
#pragma unroll
    for (int ct = 0; ct < 4; ++ct)
#pragma unroll
        for (int rt = 0; rt < 2; ++rt) oacc[ct][rt] = (v8f)(0.f);

#pragma unroll 1
    for (int j = 0; j < DFF / 128; ++j) {
        // ---- phase A: two H subtiles = A(32x512) @ W1t cols (512x16) ----
        const int ffcol = j * 128 + cw * 16 + ln;
        const _Float16* wbase = W1t + ((size_t)e * DFF + ffcol) * DM;
        __builtin_prefetch(wbase + DM * 16, 0, 1);  // W1 slab 16 ff columns ahead toward L2
        v8f hacc[2];
        hacc[0] = (v8f)(0.f); hacc[1] = (v8f)(0.f);
#pragma unroll
        for (int kb = 0; kb < 16; ++kb) {
            V16 b;
            const _Float16* bp = wbase + kb * 32 + hl * 16;
            b.h[0] = *(const v8h*)(bp);            // K contiguous 0..7
            b.h[1] = *(const v8h*)(bp + 8);        // K contiguous 8..15
#pragma unroll
            for (int rt = 0; rt < 2; ++rt) {       // B reused across row pair
                V16 a;
                const _Float16* ap = Alds + ((rg * 2 + rt) * 16 + ln) * DM + kb * 32 + hl * 16;
                a.h[0] = *(const v8h*)(ap);        // K = kb*32 + hl*16 + 0..7
                a.h[1] = *(const v8h*)(ap + 8);    // K = kb*32 + hl*16 + 8..15
                hacc[rt] = __builtin_amdgcn_wmma_f32_16x16x32_f16(
                    false, a.v, false, b.v, (short)0, hacc[rt], false, false);
            }
        }
        // bias + exact erf-GELU, write f16 H chunk to LDS
        float bias = b1[e * DFF + ffcol];
#pragma unroll
        for (int rt = 0; rt < 2; ++rt)
#pragma unroll
            for (int r = 0; r < 8; ++r) {
                float v = hacc[rt][r] + bias;
                float g = 0.5f * v * (1.f + erff(v * 0.70710678118654752f));
                Hlds[((rg * 2 + rt) * 16 + r + 8 * hl) * 128 + cw * 16 + ln] = (_Float16)g;
            }
        __syncthreads();

        // ---- phase B: out tiles += H(32x128) @ W2t rows (128x16) ----
#pragma unroll
        for (int ct = 0; ct < 4; ++ct) {
            const int col = cw * 64 + ct * 16 + ln;
            const _Float16* w2p = W2t + ((size_t)e * DM + col) * DFF + j * 128;
#pragma unroll
            for (int kb = 0; kb < 4; ++kb) {
                V16 b;
                const _Float16* bp = w2p + kb * 32 + hl * 16;
                b.h[0] = *(const v8h*)(bp);
                b.h[1] = *(const v8h*)(bp + 8);
#pragma unroll
                for (int rt = 0; rt < 2; ++rt) {   // B reused across row pair
                    V16 a;
                    const _Float16* hp = Hlds + ((rg * 2 + rt) * 16 + ln) * 128 + kb * 32 + hl * 16;
                    a.h[0] = *(const v8h*)(hp);
                    a.h[1] = *(const v8h*)(hp + 8);
                    oacc[ct][rt] = __builtin_amdgcn_wmma_f32_16x16x32_f16(
                        false, a.v, false, b.v, (short)0, oacc[ct][rt], false, false);
                }
            }
        }
        __syncthreads();                           // Hlds reused next chunk
    }

    // epilogue: out[token] += w_tok * (acc + b2)
#pragma unroll
    for (int ct = 0; ct < 4; ++ct) {
        const int col = cw * 64 + ct * 16 + ln;
        const float bb = b2[e * DM + col];
#pragma unroll
        for (int rt = 0; rt < 2; ++rt)
#pragma unroll
            for (int r = 0; r < 8; ++r) {
                int m = (rg * 2 + rt) * 16 + r + 8 * hl;   // C-layout row
                float val = swt[m] * (oacc[ct][rt][r] + bb);
                unsafeAtomicAdd(&out[(size_t)stok[m] * DM + col], val);
            }
    }
}

// ---------------- host launch --------------------------------------------------------

extern "C" void kernel_launch(void* const* d_in, const int* in_sizes, int n_in,
                              void* d_out, int out_size, void* d_ws, size_t ws_size,
                              hipStream_t stream) {
    (void)in_sizes; (void)n_in; (void)ws_size;
    const float* x  = (const float*)d_in[0];
    const float* rw = (const float*)d_in[1];
    const float* W1 = (const float*)d_in[2];
    const float* b1 = (const float*)d_in[3];
    const float* W2 = (const float*)d_in[4];
    const float* b2 = (const float*)d_in[5];
    float* out = (float*)d_out;

    // workspace layout (~42.5 MB)
    char* ws = (char*)d_ws;
    size_t o = 0;
    _Float16* xh  = (_Float16*)(ws + o); o += (size_t)NTOK * DM * 2;          //  8 MB
    _Float16* W1t = (_Float16*)(ws + o); o += (size_t)NEXP * DFF * DM * 2;    // 16 MB
    _Float16* W2t = (_Float16*)(ws + o); o += (size_t)NEXP * DM * DFF * 2;    // 16 MB
    int*   tok_list = (int*)  (ws + o);  o += (size_t)NEXP * NTOK * 4;
    float* w_list   = (float*)(ws + o);  o += (size_t)NEXP * NTOK * 4;
    int*   cnt      = (int*)  (ws + o);  o += 32;
    float* Psum     = (float*)(ws + o);  o += 32;

    (void)hipMemsetAsync(cnt, 0, 64, stream);                   // cnt + Psum
    (void)hipMemsetAsync(out, 0, sizeof(float) * (size_t)out_size, stream);

    const int nx = NTOK * DM;
    cvt_x_kernel<<<(nx + 255) / 256, 256, 0, stream>>>(x, xh, nx);
    const int nw = NEXP * DM * DFF;
    cvt_w1_kernel<<<(nw + 255) / 256, 256, 0, stream>>>(W1, W1t);
    cvt_w2_kernel<<<(nw + 255) / 256, 256, 0, stream>>>(W2, W2t);

    router_kernel<<<NTOK / 8, 256, 0, stream>>>(x, rw, tok_list, w_list, cnt, Psum);
    aux_kernel<<<1, 1, 0, stream>>>(cnt, Psum, out + (size_t)NTOK * DM);

    ffn_kernel<<<dim3(NTOK / MT, NEXP), 512, 0, stream>>>(
        xh, W1t, W2t, b1, b2, tok_list, w_list, cnt, out);
}
